// SKABlock_51049981280365
// MI455X (gfx1250) — compile-verified, matrix operand order checked by reading, not run on hardware
//
#include <hip/hip_runtime.h>
#include <hip/hip_bf16.h>

// ---------------- constants (match reference) ----------------
#define D_MODEL   1024
#define N_HEADS   16
#define HEAD_DIM  64
#define RANK      8
#define CHUNK     128
#define NCHUNK    32          // 4096 / 128
#define BATCH     2
#define SEQ       4096
#define TOKENS    (BATCH * SEQ)   // 8192
#define NLANES    (BATCH * N_HEADS)  // 32
#define RIDGE_EPS 1e-4f
#define SCALE_F   0.125f
#define DECAY_A   0.9f
#define LN_EPS    1e-5f

typedef __attribute__((ext_vector_type(16))) __bf16 v16bf;
typedef __attribute__((ext_vector_type(8)))  __bf16 v8bf;
typedef __attribute__((ext_vector_type(8)))  float  v8f;
typedef __attribute__((ext_vector_type(2)))  float  v2f;

// ---------------- LayerNorm -> bf16 ----------------
__global__ __launch_bounds__(256) void k_ln(const float* __restrict__ x,
                                            const float* __restrict__ g,
                                            const float* __restrict__ b,
                                            __bf16* __restrict__ xn) {
  const int row = blockIdx.x;
  const float* xr = x + (size_t)row * D_MODEL;
  __shared__ float red[256];
  const int tid = threadIdx.x;

  float s = 0.f;
  for (int i = tid; i < D_MODEL; i += 256) s += xr[i];
  red[tid] = s; __syncthreads();
  for (int off = 128; off > 0; off >>= 1) {
    if (tid < off) red[tid] += red[tid + off];
    __syncthreads();
  }
  const float mu = red[0] * (1.f / D_MODEL);
  __syncthreads();

  float v = 0.f;
  for (int i = tid; i < D_MODEL; i += 256) { float d = xr[i] - mu; v += d * d; }
  red[tid] = v; __syncthreads();
  for (int off = 128; off > 0; off >>= 1) {
    if (tid < off) red[tid] += red[tid + off];
    __syncthreads();
  }
  const float inv = rsqrtf(red[0] * (1.f / D_MODEL) + LN_EPS);

  for (int i = tid; i < D_MODEL; i += 256) {
    float d = (xr[i] - mu) * inv * g[i] + b[i];
    xn[(size_t)row * D_MODEL + i] = (__bf16)d;
  }
}

// ---------------- W [K][N] f32 -> Wt [N][K] bf16 ----------------
__global__ __launch_bounds__(256) void k_wcast(const float* __restrict__ W,
                                               __bf16* __restrict__ Wt) {
  int idx = blockIdx.x * 256 + threadIdx.x;
  if (idx >= D_MODEL * D_MODEL) return;
  int k = idx / D_MODEL, n = idx % D_MODEL;
  Wt[(size_t)n * D_MODEL + k] = (__bf16)W[idx];
}

// ------- bf16 fragment load per CDNA5 16-bit A/B layout (16x32 tile) -------
__device__ inline v16bf load_frag(const __bf16* __restrict__ base, int ld,
                                  int r, int kb, int lane) {
  const int rr = r + (lane & 15);
  const int kh = (lane >> 4) * 8;
  const __bf16* p = base + (size_t)rr * ld + kb + kh;
  v8bf lo = *(const v8bf*)(p);
  v8bf hi = *(const v8bf*)(p + 16);
  v16bf f;
#pragma unroll
  for (int i = 0; i < 8; ++i) { f[i] = lo[i]; f[8 + i] = hi[i]; }
  return f;
}

// ------- bf16 WMMA GEMM: C[M][N] = A[M][K] * Bt[N][K]^T (+resid) -------
__global__ __launch_bounds__(256) void k_gemm(const __bf16* __restrict__ A,
                                              const __bf16* __restrict__ Bt,
                                              const float* __restrict__ resid,
                                              float* __restrict__ C,
                                              int M, int N, int K) {
  const int lane = threadIdx.x & 31;
  const int wave = threadIdx.x >> 5;
  const int wt = blockIdx.x * 8 + wave;
  const int ntiles = N / 64;
  const int tm = wt / ntiles;
  const int tn = wt % ntiles;
  const int m0 = tm * 16;
  const int n0 = tn * 64;

  v8f acc0 = {}, acc1 = {}, acc2 = {}, acc3 = {};

  for (int kb = 0; kb < K; kb += 32) {
    v16bf a  = load_frag(A,  K, m0,      kb, lane);
    v16bf b0 = load_frag(Bt, K, n0,      kb, lane);
    v16bf b1 = load_frag(Bt, K, n0 + 16, kb, lane);
    v16bf b2 = load_frag(Bt, K, n0 + 32, kb, lane);
    v16bf b3 = load_frag(Bt, K, n0 + 48, kb, lane);
    __builtin_prefetch(A  + (size_t)(m0 + (lane & 15)) * K + kb + 32, 0, 3);
    __builtin_prefetch(Bt + (size_t)(n0 + (lane & 15)) * K + kb + 32, 0, 3);

    acc0 = __builtin_amdgcn_wmma_f32_16x16x32_bf16(false, a, false, b0, (short)0, acc0, false, false);
    acc1 = __builtin_amdgcn_wmma_f32_16x16x32_bf16(false, a, false, b1, (short)0, acc1, false, false);
    acc2 = __builtin_amdgcn_wmma_f32_16x16x32_bf16(false, a, false, b2, (short)0, acc2, false, false);
    acc3 = __builtin_amdgcn_wmma_f32_16x16x32_bf16(false, a, false, b3, (short)0, acc3, false, false);
  }

  const int n  = lane & 15;
  const int mb = m0 + ((lane >> 4) << 3);
#pragma unroll
  for (int r = 0; r < 8; ++r) {
    const int m = mb + r;
    float* cr = C + (size_t)m * N;
    float r0 = 0.f, r1 = 0.f, r2 = 0.f, r3 = 0.f;
    if (resid) {
      const float* rr = resid + (size_t)m * N;
      r0 = rr[n0 + n]; r1 = rr[n0 + 16 + n]; r2 = rr[n0 + 32 + n]; r3 = rr[n0 + 48 + n];
    }
    cr[n0 + n]      = acc0[r] + r0;
    cr[n0 + 16 + n] = acc1[r] + r1;
    cr[n0 + 32 + n] = acc2[r] + r2;
    cr[n0 + 48 + n] = acc3[r] + r3;
  }
}

// ---------------- per-(lane,chunk) ridge solve + rank-8 projection ----------
// grid: NLANES*NCHUNK blocks; writes Wr[lc][64][64]
// Phase 1: stage Kc/Vc in LDS, compute G=K^T K and RHS=K^T V with f32 WMMA
//          (waves 0-3 -> G strips, waves 4-7 -> RHS strips)
// Phase 2: LDS is re-carved for G/Wm/U/T/P; Cholesky solve + subspace QR.
__global__ __launch_bounds__(256) void k_ridge(const float* __restrict__ kk,
                                               const float* __restrict__ v,
                                               float* __restrict__ wr) {
  const int lc = blockIdx.x;
  const int ln = lc / NCHUNK;
  const int c  = lc % NCHUNK;
  const int b = ln >> 4, h = ln & 15;
  const int tid = threadIdx.x;
  const int lane = tid & 31;
  const int wave = tid >> 5;

  __shared__ __align__(16) float smem[16384];            // 64 KB
  float (*Kls)[64] = (float (*)[64])smem;                // [128][64] staging
  float (*Vls)[64] = (float (*)[64])(smem + 8192);       // [128][64] staging
  // phase-2 overlay (valid after the WMMA phase barrier)
  float (*G)[65]  = (float (*)[65])smem;                 // 4160
  float (*Wm)[65] = (float (*)[65])(smem + 4160);        // 4160
  float (*U)[8]   = (float (*)[8])(smem + 8320);         // 512
  float (*T)[9]   = (float (*)[9])(smem + 8832);         // 576
  float (*P)[64]  = (float (*)[64])(smem + 9408);        // 512
  float* red      = smem + 9920;                         // 512
  float* dots     = smem + 10432;                        // 8
  float* scal     = smem + 10440;                        // 1

  const size_t rowbase = (size_t)b * SEQ + (size_t)c * CHUNK;
  const float* Kc = kk + rowbase * D_MODEL + h * HEAD_DIM;
  const float* Vc = v  + rowbase * D_MODEL + h * HEAD_DIM;

  // ---- stage K and V chunk tiles into LDS (coalesced) ----
  for (int idx = tid; idx < CHUNK * 64; idx += 256) {
    const int t = idx >> 6, j = idx & 63;
    Kls[t][j] = Kc[(size_t)t * D_MODEL + j];
    Vls[t][j] = Vc[(size_t)t * D_MODEL + j];
  }
  __syncthreads();

  // ---- Gram products via f32 WMMA: waves 0-3: G, waves 4-7: RHS ----
  const int nn   = lane & 15;
  const int koff = (lane >> 4) * 2;          // f32 A layout K pairs
  const int m0   = (wave & 3) * 16;
  const float (*Bsrc)[64] = (wave < 4) ? Kls : Vls;

  v8f acc0 = {}, acc1 = {}, acc2 = {}, acc3 = {};
  for (int kb = 0; kb < CHUNK; kb += 4) {
    const int t = kb + koff;
    v2f a;  a[0]  = Kls[t][m0 + nn];   a[1]  = Kls[t + 1][m0 + nn];   // A = K^T
    v2f b0; b0[0] = Bsrc[t][nn];       b0[1] = Bsrc[t + 1][nn];
    v2f b1; b1[0] = Bsrc[t][16 + nn];  b1[1] = Bsrc[t + 1][16 + nn];
    v2f b2; b2[0] = Bsrc[t][32 + nn];  b2[1] = Bsrc[t + 1][32 + nn];
    v2f b3; b3[0] = Bsrc[t][48 + nn];  b3[1] = Bsrc[t + 1][48 + nn];
    acc0 = __builtin_amdgcn_wmma_f32_16x16x4_f32(false, a, false, b0, (short)0, acc0, false, false);
    acc1 = __builtin_amdgcn_wmma_f32_16x16x4_f32(false, a, false, b1, (short)0, acc1, false, false);
    acc2 = __builtin_amdgcn_wmma_f32_16x16x4_f32(false, a, false, b2, (short)0, acc2, false, false);
    acc3 = __builtin_amdgcn_wmma_f32_16x16x4_f32(false, a, false, b3, (short)0, acc3, false, false);
  }
  __syncthreads();   // all waves done reading Kls/Vls; safe to overlay

  // scatter accumulators into G (with ridge eps on diagonal) / Wm
  {
    float (*dst)[65] = (wave < 4) ? G : Wm;
    const bool isG = (wave < 4);
    const int mb = m0 + ((lane >> 4) << 3);
#pragma unroll
    for (int r = 0; r < 8; ++r) {
      const int m = mb + r;
      dst[m][nn]      = acc0[r] + ((isG && m == nn)      ? RIDGE_EPS : 0.f);
      dst[m][16 + nn] = acc1[r] + ((isG && m == 16 + nn) ? RIDGE_EPS : 0.f);
      dst[m][32 + nn] = acc2[r] + ((isG && m == 32 + nn) ? RIDGE_EPS : 0.f);
      dst[m][48 + nn] = acc3[r] + ((isG && m == 48 + nn) ? RIDGE_EPS : 0.f);
    }
  }
  __syncthreads();

  // ---- in-place Cholesky (lower) of G ----
  for (int j = 0; j < 64; ++j) {
    if (tid == 0) G[j][j] = sqrtf(G[j][j]);
    __syncthreads();
    if (tid < 64 && tid > j) G[tid][j] /= G[j][j];
    __syncthreads();
    if (tid < 64 && tid > j) {
      const float lij = G[tid][j];
      for (int kc = j + 1; kc <= tid; ++kc) G[tid][kc] -= lij * G[kc][j];
    }
    __syncthreads();
  }

  // ---- solve G W = RHS, one RHS column per thread ----
  if (tid < 64) {
    const int cc = tid;
    for (int i = 0; i < 64; ++i) {
      float s = Wm[i][cc];
      for (int kc = 0; kc < i; ++kc) s -= G[i][kc] * Wm[kc][cc];
      Wm[i][cc] = s / G[i][i];
    }
    for (int i = 63; i >= 0; --i) {
      float s = Wm[i][cc];
      for (int kc = i + 1; kc < 64; ++kc) s -= G[kc][i] * Wm[kc][cc];
      Wm[i][cc] = s / G[i][i];
    }
  }
  __syncthreads();

  // ---- subspace iteration: U = I(:,0:8); twice: U = qr(W U).Q (MGS) ----
  for (int idx = tid; idx < 512; idx += 256) {
    const int i = idx >> 3, cc = idx & 7;
    U[i][cc] = (i == cc) ? 1.f : 0.f;
  }
  __syncthreads();
  for (int p = 0; p < 2; ++p) {
    for (int idx = tid; idx < 512; idx += 256) {
      const int i = idx >> 3, cc = idx & 7;
      float s = 0.f;
      for (int kc = 0; kc < 64; ++kc) s += Wm[i][kc] * U[kc][cc];
      T[i][cc] = s;
    }
    __syncthreads();
    for (int cc = 0; cc < 8; ++cc) {
      if (tid < 64) red[tid] = T[tid][cc] * T[tid][cc];
      __syncthreads();
      if (tid == 0) {
        float s = 0.f;
        for (int t2 = 0; t2 < 64; ++t2) s += red[t2];
        *scal = rsqrtf(fmaxf(s, 1e-30f));
      }
      __syncthreads();
      if (tid < 64) U[tid][cc] = T[tid][cc] * (*scal);
      __syncthreads();
      if (tid < 64)
        for (int c2 = cc + 1; c2 < 8; ++c2) red[(c2 << 6) + tid] = U[tid][cc] * T[tid][c2];
      __syncthreads();
      if (tid < 8 && tid > cc) {
        float s = 0.f;
        for (int t2 = 0; t2 < 64; ++t2) s += red[(tid << 6) + t2];
        dots[tid] = s;
      }
      __syncthreads();
      if (tid < 64)
        for (int c2 = cc + 1; c2 < 8; ++c2) T[tid][c2] -= dots[c2] * U[tid][cc];
      __syncthreads();
    }
  }

  // ---- P = U^T W (8 x 64) ----
  for (int idx = tid; idx < 512; idx += 256) {
    const int r = idx >> 6, j = idx & 63;
    float s = 0.f;
    for (int t2 = 0; t2 < 64; ++t2) s += U[t2][r] * Wm[t2][j];
    P[r][j] = s;
  }
  __syncthreads();

  // ---- Wr = U P -> global ----
  float* wrg = wr + (size_t)lc * 4096;
  for (int idx = tid; idx < 4096; idx += 256) {
    const int i = idx >> 6, j = idx & 63;
    float s = 0.f;
#pragma unroll
    for (int r = 0; r < RANK; ++r) s += U[i][r] * P[r][j];
    wrg[idx] = s;
  }
}

// ---------------- decay prefix over chunks: Wr -> S_c (in place) ------------
__global__ __launch_bounds__(256) void k_decay(float* __restrict__ wr) {
  const int ln = blockIdx.x;
  const int tid = threadIdx.x;
  float s[16];
#pragma unroll
  for (int e = 0; e < 16; ++e) s[e] = 0.f;
  for (int c = 0; c < NCHUNK; ++c) {
    float* p = wr + ((size_t)ln * NCHUNK + c) * 4096;
#pragma unroll
    for (int e = 0; e < 16; ++e) {
      const int idx = e * 256 + tid;
      s[e] = DECAY_A * s[e] + p[idx];
      p[idx] = s[e];
    }
  }
}

// ---------------- out = SCALE * Qc @ S_c via f32 WMMA (16x16x4) ------------
__global__ __launch_bounds__(256) void k_apply(const float* __restrict__ q,
                                               const float* __restrict__ sall,
                                               __bf16* __restrict__ o) {
  const int lc = blockIdx.x;
  const int ln = lc / NCHUNK;
  const int c  = lc % NCHUNK;
  const int b = ln >> 4, h = ln & 15;
  const int tid = threadIdx.x;

  __shared__ float Ssh[64 * 64];
  const float* sg = sall + (size_t)lc * 4096;
  for (int i = tid; i < 4096; i += 256) Ssh[i] = sg[i];
  __syncthreads();

  const int lane = tid & 31;
  const int wave = tid >> 5;
  const size_t rowbase = (size_t)b * SEQ + (size_t)c * CHUNK;
  const float* Qc = q + rowbase * D_MODEL + h * HEAD_DIM;

  const int m0   = wave * 16;
  const int nn   = lane & 15;
  const int koff = (lane >> 4) * 2;

  v8f acc0 = {}, acc1 = {}, acc2 = {}, acc3 = {};

  for (int kb = 0; kb < 64; kb += 4) {
    const float* qp = Qc + (size_t)(m0 + nn) * D_MODEL + kb + koff;
    v2f a; a[0] = qp[0]; a[1] = qp[1];
    const int kr = (kb + koff) * 64;
    v2f b0, b1, b2, b3;
    b0[0] = Ssh[kr + nn];        b0[1] = Ssh[kr + 64 + nn];
    b1[0] = Ssh[kr + 16 + nn];   b1[1] = Ssh[kr + 64 + 16 + nn];
    b2[0] = Ssh[kr + 32 + nn];   b2[1] = Ssh[kr + 64 + 32 + nn];
    b3[0] = Ssh[kr + 48 + nn];   b3[1] = Ssh[kr + 64 + 48 + nn];
    acc0 = __builtin_amdgcn_wmma_f32_16x16x4_f32(false, a, false, b0, (short)0, acc0, false, false);
    acc1 = __builtin_amdgcn_wmma_f32_16x16x4_f32(false, a, false, b1, (short)0, acc1, false, false);
    acc2 = __builtin_amdgcn_wmma_f32_16x16x4_f32(false, a, false, b2, (short)0, acc2, false, false);
    acc3 = __builtin_amdgcn_wmma_f32_16x16x4_f32(false, a, false, b3, (short)0, acc3, false, false);
  }

  const int mb = m0 + ((lane >> 4) << 3);
#pragma unroll
  for (int r = 0; r < 8; ++r) {
    const int m = mb + r;
    __bf16* orow = o + (rowbase + m) * D_MODEL + h * HEAD_DIM;
    orow[nn]      = (__bf16)(SCALE_F * acc0[r]);
    orow[16 + nn] = (__bf16)(SCALE_F * acc1[r]);
    orow[32 + nn] = (__bf16)(SCALE_F * acc2[r]);
    orow[48 + nn] = (__bf16)(SCALE_F * acc3[r]);
  }
}

// ---------------- host-side orchestration ----------------
extern "C" void kernel_launch(void* const* d_in, const int* in_sizes, int n_in,
                              void* d_out, int out_size, void* d_ws, size_t ws_size,
                              hipStream_t stream) {
  const float* x   = (const float*)d_in[0];
  const float* lng = (const float*)d_in[1];
  const float* lnb = (const float*)d_in[2];
  const float* Wq  = (const float*)d_in[3];
  const float* Wk  = (const float*)d_in[4];
  const float* Wv  = (const float*)d_in[5];
  const float* Wo  = (const float*)d_in[6];
  float* out = (float*)d_out;

  char* ws = (char*)d_ws;
  size_t off = 0;
  auto alloc = [&](size_t bytes) -> void* {
    void* p = ws + off;
    off += (bytes + 255) & ~(size_t)255;
    return p;
  };

  __bf16* xn  = (__bf16*)alloc((size_t)TOKENS * D_MODEL * 2);
  __bf16* Wtq = (__bf16*)alloc((size_t)D_MODEL * D_MODEL * 2);
  __bf16* Wtk = (__bf16*)alloc((size_t)D_MODEL * D_MODEL * 2);
  __bf16* Wtv = (__bf16*)alloc((size_t)D_MODEL * D_MODEL * 2);
  __bf16* Wto = (__bf16*)alloc((size_t)D_MODEL * D_MODEL * 2);
  float*  qf  = (float*)alloc((size_t)TOKENS * D_MODEL * 4);
  float*  kf  = (float*)alloc((size_t)TOKENS * D_MODEL * 4);
  float*  vf  = (float*)alloc((size_t)TOKENS * D_MODEL * 4);
  float*  wr  = (float*)alloc((size_t)NLANES * NCHUNK * 4096 * 4);
  __bf16* obf = (__bf16*)alloc((size_t)TOKENS * D_MODEL * 2);

  // 1. LayerNorm -> bf16
  k_ln<<<TOKENS, 256, 0, stream>>>(x, lng, lnb, xn);

  // 2. weight transpose+cast
  const int wblocks = (D_MODEL * D_MODEL + 255) / 256;
  k_wcast<<<wblocks, 256, 0, stream>>>(Wq, Wtq);
  k_wcast<<<wblocks, 256, 0, stream>>>(Wk, Wtk);
  k_wcast<<<wblocks, 256, 0, stream>>>(Wv, Wtv);
  k_wcast<<<wblocks, 256, 0, stream>>>(Wo, Wto);

  // 3. QKV GEMMs (bf16 WMMA, f32 accumulate)
  const int gblocks = (TOKENS / 16) * (D_MODEL / 64) / 8;   // 1024
  k_gemm<<<gblocks, 256, 0, stream>>>(xn, Wtq, nullptr, qf, TOKENS, D_MODEL, D_MODEL);
  k_gemm<<<gblocks, 256, 0, stream>>>(xn, Wtk, nullptr, kf, TOKENS, D_MODEL, D_MODEL);
  k_gemm<<<gblocks, 256, 0, stream>>>(xn, Wtv, nullptr, vf, TOKENS, D_MODEL, D_MODEL);

  // 4a. per-(lane,chunk) ridge + rank-8 projection -> Wr  (fully parallel, WMMA Gram)
  k_ridge<<<NLANES * NCHUNK, 256, 0, stream>>>(kf, vf, wr);
  // 4b. decayed prefix over chunks: Wr -> S_c (in place)
  k_decay<<<NLANES, 256, 0, stream>>>(wr);
  // 4c. out = SCALE * Qc @ S_c  (f32 WMMA) -> bf16
  k_apply<<<NLANES * NCHUNK, 256, 0, stream>>>(qf, wr, obf);

  // 5. output GEMM + residual
  k_gemm<<<gblocks, 256, 0, stream>>>(obf, Wto, x, out, TOKENS, D_MODEL, D_MODEL);
}
